// GNNEncoder_45062796870437
// MI455X (gfx1250) — matrix operand order repaired and checked by reading, not compiled, not run on hardware
//
#include <hip/hip_runtime.h>

// ---------------------------------------------------------------------------
// DGCNN / DynamicEdgeConv encoder for MI455X (gfx1250, wave32, WMMA).
// B=8, N=2048, K=16, DIMS = [3,32,128,256,64,32,16].
//   kNN Gram tiles   : V_WMMA_F32_16X16X4_F32 (fp32, preserves top-k ordering)
//   edge-MLP GEMMs   : V_WMMA_F32_16X16X32_F16 (fp32 accumulate)
//   16x2048 distance block staged in LDS (CDNA5 320KB/WGP)
// ---------------------------------------------------------------------------

typedef _Float16 h16;
typedef __attribute__((ext_vector_type(16))) _Float16 v16h;
typedef __attribute__((ext_vector_type(8)))  _Float16 v8h;
typedef __attribute__((ext_vector_type(8)))  float    v8f;
typedef __attribute__((ext_vector_type(4)))  float    v4f;
typedef __attribute__((ext_vector_type(2)))  float    v2f;

#define BATCH 8
#define NPTS  2048
#define KNN   16

// 16-bit A 16x32 layout (contiguous, matches B): per lane, halves 0..15 =
// K=kb..kb+15, kb = k0 + 16*(lane>>4). p = &row[k0+16*hi].
__device__ inline v16h load_a16(const h16* p) {
    return *(const v16h*)p;
}

// ---------------------------------------------------------------------------
// Kernel 0: pad layer-0 input [B*N,3] -> [B*N,4] (zero 4th channel).
// ---------------------------------------------------------------------------
__global__ void pad_kernel(const float* __restrict__ x, float* __restrict__ xpad, int total) {
    int t = blockIdx.x * blockDim.x + threadIdx.x;
    if (t >= total) return;
    const float* p = x + (size_t)t * 3;
    float* q = xpad + (size_t)t * 4;
    q[0] = p[0]; q[1] = p[1]; q[2] = p[2]; q[3] = 0.f;
}

// ---------------------------------------------------------------------------
// Kernel 1: squared norms over stride-Dpad rows (pad channels are zero).
// ---------------------------------------------------------------------------
__global__ void sqnorm_kernel(const float* __restrict__ x, float* __restrict__ sq,
                              int Dpad, int total) {
    int t = blockIdx.x * blockDim.x + threadIdx.x;
    if (t >= total) return;
    const float* p = x + (size_t)t * Dpad;
    float s = 0.f;
    for (int c = 0; c < Dpad; ++c) { float v = p[c]; s += v * v; }
    sq[t] = s;
}

// ---------------------------------------------------------------------------
// Kernel 2: weight convert: W1 [K1=2din][N1=2dout] fp32 -> w1t f16 [N1][Kpad]
//           (zero-padded in K), W2 [N1][N2=dout] fp32 -> w2t f16 [N2][N1].
// ---------------------------------------------------------------------------
__global__ void convw_kernel(const float* __restrict__ W1, const float* __restrict__ W2,
                             h16* __restrict__ w1t, h16* __restrict__ w2t,
                             int K1, int N1, int Kpad, int N2) {
    int t = blockIdx.x * blockDim.x + threadIdx.x;
    int t1 = N1 * Kpad;
    if (t < t1) {
        int n = t / Kpad, k = t % Kpad;
        w1t[t] = (k < K1) ? (h16)W1[k * N1 + n] : (h16)0.f;
    } else {
        int t2 = t - t1;
        if (t2 < N2 * N1) {
            int n = t2 / N1, k = t2 % N1;
            w2t[t2] = (h16)W2[k * N2 + n];
        }
    }
}

// ---------------------------------------------------------------------------
// Kernel 3: fused kNN. One block = 16 query rows of one batch.
// Input rows are stride-Dpad and zero-padded => branch-free v2f loads.
// dist stored column-major dist[j*16 + m] so each 16x16 tile is written
// with two contiguous v4f LDS stores per lane.
// dyn LDS: dist[N*16] | xiL[16*Dpad] | sqiL[16]
// ---------------------------------------------------------------------------
__global__ void knn_kernel(const float* __restrict__ x, const float* __restrict__ sq,
                           int* __restrict__ nbr, int Dpad, int N) {
    extern __shared__ float kls[];
    float* dist = kls;                 // N * 16, [j][m]
    float* xiL  = kls + N * 16;        // 16 * Dpad
    float* sqiL = xiL + 16 * Dpad;     // 16

    const int tid  = threadIdx.x;
    const int w    = tid >> 5;
    const int lane = tid & 31;
    const int lo   = lane & 15;
    const int hi   = lane >> 4;

    const int ntiles = N / 16;
    const int b  = blockIdx.x / ntiles;
    const int it = blockIdx.x % ntiles;
    const int i0 = it * 16;
    const float* xb = x + (size_t)b * N * Dpad;

    // stage 16 query rows: contiguous 16*Dpad floats
    for (int t = tid; t < 16 * Dpad; t += blockDim.x)
        xiL[t] = xb[(size_t)i0 * Dpad + t];
    if (tid < 16) sqiL[tid] = sq[(size_t)b * N + i0 + tid];
    __syncthreads();

    // distance tiles: wave w handles j-tiles w, w+8, ...
    for (int jt = w; jt < ntiles; jt += 8) {
        const int j = jt * 16 + lo;
        const float* xjp = xb + (size_t)j * Dpad;
        v8f acc;
#pragma unroll
        for (int r = 0; r < 8; ++r) acc[r] = 0.f;
        for (int k0 = 0; k0 < Dpad; k0 += 4) {
            const int kk = k0 + 2 * hi;
            v2f A, Bv;
            A[0] = xiL[lo * Dpad + kk];
            A[1] = xiL[lo * Dpad + kk + 1];
            Bv = *(const v2f*)(xjp + kk);                 // unconditional b64 load
            acc = __builtin_amdgcn_wmma_f32_16x16x4_f32(
                      false, A, false, Bv, (short)0, acc, false, false);
        }
        const float sqj = sq[(size_t)b * N + j];
        v4f d0, d1;
#pragma unroll
        for (int r = 0; r < 4; ++r) d0[r] = sqiL[r + 8 * hi]     + sqj - 2.f * acc[r];
#pragma unroll
        for (int r = 0; r < 4; ++r) d1[r] = sqiL[r + 4 + 8 * hi] + sqj - 2.f * acc[r + 4];
        float* dp = dist + (size_t)j * 16 + 8 * hi;
        *(v4f*)dp       = d0;                             // two b128 LDS stores
        *(v4f*)(dp + 4) = d1;
    }
    __syncthreads();

    // top-16 selection: wave w owns rows 2w, 2w+1 (m index)
    for (int rr = 0; rr < 2; ++rr) {
        const int rowm = w * 2 + rr;
        for (int pass = 0; pass < KNN; ++pass) {
            unsigned long long best = ~0ull;
            for (int t = lane; t < N; t += 32) {
                unsigned u = __float_as_uint(dist[(size_t)t * 16 + rowm]);
                u = (u & 0x80000000u) ? ~u : (u | 0x80000000u);   // sortable float
                unsigned long long k64 = ((unsigned long long)u << 32) | (unsigned)t;
                if (k64 < best) best = k64;
            }
#pragma unroll
            for (int off = 16; off >= 1; off >>= 1) {
                unsigned long long o = __shfl_xor(best, off);
                if (o < best) best = o;
            }
            const int jmin = (int)(best & 0xffffffffu);
            if (lane == 0) {
                nbr[((size_t)b * N + i0 + rowm) * KNN + pass] = jmin;
                dist[(size_t)jmin * 16 + rowm] = __int_as_float(0x7f7fffff); // FLT_MAX
            }
        }
    }
}

// ---------------------------------------------------------------------------
// Kernel 4: fused EdgeConv layer. Block = 8 nodes, one wave per node
// (its 16 edges are exactly one M=16 WMMA tile). N-tiles processed in
// pairs: one A-tile LDS load feeds two WMMAs.
// dyn LDS: eLds[128*Kpad] | hLds[128*N1] halves
// ---------------------------------------------------------------------------
__global__ void edgeconv_kernel(const float* __restrict__ xin, const int* __restrict__ nbr,
                                const h16* __restrict__ w1t, const float* __restrict__ b1,
                                const h16* __restrict__ w2t, const float* __restrict__ b2,
                                float* __restrict__ xout,
                                int din, int Dpad, int dout, int Kpad, int N) {
    extern __shared__ h16 els[];
    const int N1 = 2 * dout;           // GEMM1 output width == GEMM2 K
    h16* eLds = els;                   // 128 * Kpad
    h16* hLds = els + 128 * Kpad;      // 128 * N1

    const int tid  = threadIdx.x;
    const int w    = tid >> 5;
    const int lane = tid & 31;
    const int lo   = lane & 15;
    const int hi   = lane >> 4;

    const int node = blockIdx.x * 8 + w;      // global node id = b*N + ni
    const int b    = node / N;
    const float* xi = xin + (size_t)node * Dpad;

    // prefetch f16 weights into near caches (gfx1250 global_prefetch_b8)
    {
        const int w1b = N1 * Kpad * (int)sizeof(h16);
        const int w2b = dout * N1 * (int)sizeof(h16);
        for (int o = tid * 256; o < w1b; o += 256 * 256)
            __builtin_prefetch((const char*)w1t + o, 0, 1);
        for (int o = tid * 256; o < w2b; o += 256 * 256)
            __builtin_prefetch((const char*)w2t + o, 0, 1);
    }

    // gather edge features [x_i, x_j - x_i] for 16 neighbors into LDS (f16)
    for (int s = 0; s < KNN; ++s) {
        const int j = nbr[(size_t)node * KNN + s];
        const float* xj = xin + ((size_t)b * N + j) * Dpad;
        h16* er = eLds + (w * 16 + s) * Kpad;
        for (int c = lane; c < din; c += 32) {
            float a = xi[c];
            er[c]       = (h16)a;
            er[din + c] = (h16)(xj[c] - a);
        }
        for (int c = 2 * din + lane; c < Kpad; c += 32)
            er[c] = (h16)0.f;
    }

    // ---- GEMM1: [16 x Kpad] x [Kpad x N1] + b1, relu -> hLds f16 ----
    // N1 is always a multiple of 32: pure pair loop.
    for (int n0 = 0; n0 < N1; n0 += 32) {
        const float bias0 = b1[n0 + lo];
        const float bias1 = b1[n0 + 16 + lo];
        v8f acc0, acc1;
#pragma unroll
        for (int r = 0; r < 8; ++r) { acc0[r] = bias0; acc1[r] = bias1; }
        const h16* ab = eLds + (w * 16 + lo) * Kpad + 16 * hi;
        const h16* bb0 = w1t + (size_t)(n0 + lo) * Kpad + 16 * hi;
        const h16* bb1 = w1t + (size_t)(n0 + 16 + lo) * Kpad + 16 * hi;
        for (int k0 = 0; k0 < Kpad; k0 += 32) {
            v16h A  = load_a16(ab + k0);
            v16h B0 = *(const v16h*)(bb0 + k0);
            v16h B1 = *(const v16h*)(bb1 + k0);
            acc0 = __builtin_amdgcn_wmma_f32_16x16x32_f16(
                       false, A, false, B0, (short)0, acc0, false, false);
            acc1 = __builtin_amdgcn_wmma_f32_16x16x32_f16(
                       false, A, false, B1, (short)0, acc1, false, false);
        }
#pragma unroll
        for (int r = 0; r < 8; ++r) {
            const int m = r + 8 * hi;
            float v0 = acc0[r]; v0 = v0 > 0.f ? v0 : 0.f;
            float v1 = acc1[r]; v1 = v1 > 0.f ? v1 : 0.f;
            hLds[(w * 16 + m) * N1 + (n0 + lo)]      = (h16)v0;
            hLds[(w * 16 + m) * N1 + (n0 + 16 + lo)] = (h16)v1;
        }
    }

    // ---- GEMM2: [16 x N1] x [N1 x dout] + b2, relu, mean over 16 edges ----
    const h16* ab2 = hLds + (w * 16 + lo) * N1 + 16 * hi;
    int n0 = 0;
    for (; n0 + 32 <= dout; n0 += 32) {
        const float bias0 = b2[n0 + lo];
        const float bias1 = b2[n0 + 16 + lo];
        v8f acc0, acc1;
#pragma unroll
        for (int r = 0; r < 8; ++r) { acc0[r] = bias0; acc1[r] = bias1; }
        const h16* bb0 = w2t + (size_t)(n0 + lo) * N1 + 16 * hi;
        const h16* bb1 = w2t + (size_t)(n0 + 16 + lo) * N1 + 16 * hi;
        for (int k0 = 0; k0 < N1; k0 += 32) {
            v16h A  = load_a16(ab2 + k0);
            v16h B0 = *(const v16h*)(bb0 + k0);
            v16h B1 = *(const v16h*)(bb1 + k0);
            acc0 = __builtin_amdgcn_wmma_f32_16x16x32_f16(
                       false, A, false, B0, (short)0, acc0, false, false);
            acc1 = __builtin_amdgcn_wmma_f32_16x16x32_f16(
                       false, A, false, B1, (short)0, acc1, false, false);
        }
        float s0 = 0.f, s1 = 0.f;
#pragma unroll
        for (int r = 0; r < 8; ++r) {
            float v0 = acc0[r]; s0 += (v0 > 0.f ? v0 : 0.f);   // relu per edge
            float v1 = acc1[r]; s1 += (v1 > 0.f ? v1 : 0.f);
        }
        s0 += __shfl_xor(s0, 16);                  // combine M halves
        s1 += __shfl_xor(s1, 16);
        float o0 = s0 * (1.0f / (float)KNN); o0 = o0 > 0.f ? o0 : 0.f;
        float o1 = s1 * (1.0f / (float)KNN); o1 = o1 > 0.f ? o1 : 0.f;
        if (hi == 0) {
            xout[(size_t)node * dout + n0 + lo]      = o0;
            xout[(size_t)node * dout + n0 + 16 + lo] = o1;
        }
    }
    if (n0 < dout) {                               // single-tile tail (dout==16)
        const float bias = b2[n0 + lo];
        v8f acc;
#pragma unroll
        for (int r = 0; r < 8; ++r) acc[r] = bias;
        const h16* bb = w2t + (size_t)(n0 + lo) * N1 + 16 * hi;
        for (int k0 = 0; k0 < N1; k0 += 32) {
            v16h A = load_a16(ab2 + k0);
            v16h B = *(const v16h*)(bb + k0);
            acc = __builtin_amdgcn_wmma_f32_16x16x32_f16(
                      false, A, false, B, (short)0, acc, false, false);
        }
        float s = 0.f;
#pragma unroll
        for (int r = 0; r < 8; ++r) { float v = acc[r]; s += (v > 0.f ? v : 0.f); }
        s += __shfl_xor(s, 16);
        float o = s * (1.0f / (float)KNN); o = o > 0.f ? o : 0.f;
        if (hi == 0) xout[(size_t)node * dout + n0 + lo] = o;
    }
}

// ---------------------------------------------------------------------------
// Kernel 5: global mean pool over N then [16 x 2] head.
// ---------------------------------------------------------------------------
__global__ void pool_kernel(const float* __restrict__ xf, const float* __restrict__ Wf,
                            const float* __restrict__ bf, float* __restrict__ out, int N) {
    __shared__ float sm[256];
    const int b = blockIdx.x;
    const int t = threadIdx.x;
    const int c = t & 15;
    const int g = t >> 4;
    float s = 0.f;
    for (int n = g; n < N; n += 16) s += xf[((size_t)b * N + n) * 16 + c];
    sm[t] = s;
    __syncthreads();
    if (t < 16) {
        float tot = 0.f;
        for (int gg = 0; gg < 16; ++gg) tot += sm[gg * 16 + t];
        sm[t] = tot / (float)N;
    }
    __syncthreads();
    if (t < 2) {
        float o = bf[t];
        for (int cc = 0; cc < 16; ++cc) o += sm[cc] * Wf[cc * 2 + t];
        out[b * 2 + t] = o;
    }
}

// ---------------------------------------------------------------------------
extern "C" void kernel_launch(void* const* d_in, const int* in_sizes, int n_in,
                              void* d_out, int out_size, void* d_ws, size_t ws_size,
                              hipStream_t stream) {
    (void)in_sizes; (void)n_in; (void)out_size; (void)ws_size;
    const int B = BATCH, N = NPTS;
    static const int DIMS[7] = {3, 32, 128, 256, 64, 32, 16};

    char* ws = (char*)d_ws;
    size_t off = 0;
    auto walloc = [&](size_t bytes) -> void* {
        void* p = ws + off;
        off = (off + bytes + 255) & ~(size_t)255;
        return p;
    };

    float* xa    = (float*)walloc((size_t)B * N * 256 * sizeof(float));
    float* xb2   = (float*)walloc((size_t)B * N * 256 * sizeof(float));
    float* xpad0 = (float*)walloc((size_t)B * N * 4 * sizeof(float));
    float* sq    = (float*)walloc((size_t)B * N * sizeof(float));
    int*   d_nbr = (int*)  walloc((size_t)B * N * KNN * sizeof(int));

    h16* w1t[6]; h16* w2t[6]; int KpadL[6];
    for (int l = 0; l < 6; ++l) {
        const int din = DIMS[l], dout = DIMS[l + 1];
        const int N1 = 2 * dout;
        const int Kpad = (2 * din + 31) & ~31;
        KpadL[l] = Kpad;
        w1t[l] = (h16*)walloc((size_t)N1 * Kpad * sizeof(h16));
        w2t[l] = (h16*)walloc((size_t)dout * N1 * sizeof(h16));
    }

    // weight convert/transpose to f16 (deterministic; re-done every call)
    for (int l = 0; l < 6; ++l) {
        const int din = DIMS[l], dout = DIMS[l + 1];
        const int N1 = 2 * dout, Kpad = KpadL[l];
        const int elems = N1 * Kpad + dout * N1;
        convw_kernel<<<(elems + 255) / 256, 256, 0, stream>>>(
            (const float*)d_in[1 + 4 * l], (const float*)d_in[3 + 4 * l],
            w1t[l], w2t[l], 2 * din, N1, Kpad, dout);
    }

    // pad layer-0 input to 4 channels
    pad_kernel<<<(B * N + 255) / 256, 256, 0, stream>>>((const float*)d_in[0], xpad0, B * N);

    const float* xcur = xpad0;
    float* bufs[2] = {xa, xb2};
    for (int l = 0; l < 6; ++l) {
        const int din = DIMS[l], dout = DIMS[l + 1];
        const int Dpad = (l == 0) ? 4 : din;    // row stride of xcur (padded)
        const int Kpad = KpadL[l];

        sqnorm_kernel<<<(B * N + 255) / 256, 256, 0, stream>>>(xcur, sq, Dpad, B * N);

        const size_t knnLds = (size_t)N * 16 * 4 + (size_t)16 * Dpad * 4 + 64;
        knn_kernel<<<B * (N / 16), 256, knnLds, stream>>>(xcur, sq, d_nbr, Dpad, N);

        float* xnext = bufs[l & 1];
        const size_t ecLds = ((size_t)128 * Kpad + (size_t)128 * 2 * dout) * sizeof(h16);
        edgeconv_kernel<<<B * N / 8, 256, ecLds, stream>>>(
            xcur, d_nbr, w1t[l], (const float*)d_in[2 + 4 * l],
            w2t[l], (const float*)d_in[4 + 4 * l], xnext, din, Dpad, dout, Kpad, N);

        xcur = xnext;
    }

    pool_kernel<<<B, 256, 0, stream>>>(xcur, (const float*)d_in[25],
                                       (const float*)d_in[26], (float*)d_out, N);
}
